// EncoderCell_21534966022552
// MI455X (gfx1250) — matrix operand order repaired and checked by reading, not compiled, not run on hardware
//
#include <hip/hip_runtime.h>
#include <math.h>

// ---------------- problem dims ----------------
#define B_   32
#define T_   49
#define NB   24
#define H_   128
#define G_   9
#define ROWS (B_*NB)        // 768
#define KTOT 896            // 7*H  (h | h_tb | p | h_ta | x_s | g_t | g_s)
#define GT_  (G_*T_)        // 441
#define LBPAD 40            // bf16 elems per LDS row (32 data + 8 pad) -> conflict-free b128

typedef __bf16 bf16;
typedef __attribute__((ext_vector_type(16))) __bf16 v16bf;
typedef __attribute__((ext_vector_type(8)))  float  v8f;
typedef __attribute__((ext_vector_type(4)))  int    v4i;

__device__ __forceinline__ float sigf(float x) { return 1.f / (1.f + __expf(-x)); }

// ---- async global->LDS staging (CDNA5 GLOBAL_LOAD_ASYNC_TO_LDS_B128, ASYNCcnt) ----
#if defined(__has_builtin)
#if __has_builtin(__builtin_amdgcn_global_load_async_to_lds_b128)
#define HAVE_ASYNC_LDS 1
#endif
#endif

__device__ __forceinline__ void stage16(bf16* ldst, const bf16* gsrc)
{
#ifdef HAVE_ASYNC_LDS
    __builtin_amdgcn_global_load_async_to_lds_b128(
        (__attribute__((address_space(1))) v4i*)gsrc,
        (__attribute__((address_space(3))) v4i*)ldst, 0, 0);
#else
    *(uint4*)ldst = *(const uint4*)gsrc;
#endif
}

__device__ __forceinline__ void stage_commit()
{
#ifdef HAVE_ASYNC_LDS
    asm volatile("s_wait_asynccnt 0x0" ::: "memory");
#endif
    __syncthreads();
}

// ---------------- kernel 1: pack activations into bf16 A [T][768][896] ----------------
__global__ __launch_bounds__(256) void pack_A(const float* __restrict__ h,
                                              const float* __restrict__ p,
                                              const float* __restrict__ gt,
                                              const float* __restrict__ gs,
                                              bf16* __restrict__ A)
{
    int i = blockIdx.x * 256 + threadIdx.x;          // over T*ROWS*H
    if (i >= T_ * ROWS * H_) return;
    int hh  = i & (H_ - 1);
    int row = (i >> 7) % ROWS;
    int t   = i / (ROWS * H_);
    int b = row / NB, nb = row % NB;
    size_t base = ((size_t)(b * T_ + t) * NB + nb) * H_ + hh;   // [B,T,NB,H]
    float vh  = h[base];
    float vtb = (t > 0)      ? h[base - (size_t)NB * H_] : 0.f;
    float vta = (t < T_ - 1) ? h[base + (size_t)NB * H_] : 0.f;
    float vp  = p[base];
    float vs  = ((nb > 0)      ? h[base - H_] : 0.f)
              + ((nb < NB - 1) ? h[base + H_] : 0.f);
    float vgt = gt[(size_t)row * H_ + hh];
    float vgs = gs[((size_t)b * T_ + t) * H_ + hh];
    bf16* arow = A + ((size_t)t * ROWS + row) * KTOT;
    arow[0 * H_ + hh] = (bf16)vh;
    arow[1 * H_ + hh] = (bf16)vtb;
    arow[2 * H_ + hh] = (bf16)vp;
    arow[3 * H_ + hh] = (bf16)vta;
    arow[4 * H_ + hh] = (bf16)vs;
    arow[5 * H_ + hh] = (bf16)vgt;
    arow[6 * H_ + hh] = (bf16)vgs;
}

// ---------------- kernel 2: tiled transpose-pack  f32 [slab][K][128] -> bf16 [slab][128][ktot] ----------------
__global__ __launch_bounds__(256) void pack_wT(const float* __restrict__ src,
                                               bf16* __restrict__ dst,
                                               int ksrc, int ktot, int koff)
{
    __shared__ float tile[32][33];
    int slab = blockIdx.z;
    int k0 = blockIdx.x * 32, n0 = blockIdx.y * 32;
    int tx = threadIdx.x, ty = threadIdx.y;          // 32 x 8
    const float* s = src + (size_t)slab * ksrc * H_;
#pragma unroll
    for (int i = 0; i < 4; i++)
        tile[ty + i * 8][tx] = s[(size_t)(k0 + ty + i * 8) * H_ + n0 + tx];
    __syncthreads();
    bf16* d = dst + (size_t)slab * H_ * ktot;
#pragma unroll
    for (int i = 0; i < 4; i++)
        d[(size_t)(n0 + ty + i * 8) * ktot + koff + k0 + tx] = (bf16)tile[tx][ty + i * 8];
}

// ---------------- kernel 3: init recurrence state ----------------
__global__ __launch_bounds__(256) void init_gt(const float* __restrict__ g0,
                                               const float* __restrict__ c0,
                                               bf16* __restrict__ gbf,
                                               float* __restrict__ cbuf)
{
    int i = blockIdx.x * 256 + threadIdx.x;
    if (i < ROWS * H_) { gbf[i] = (bf16)g0[i]; cbuf[i] = c0[i]; }
}

// ---------------- kernel 4: main WMMA GEMM  pre[gt][768][128] = A[t] x WpT[gt] + bias ----------------
__global__ __launch_bounds__(256) void gemm_pre(const bf16* __restrict__ A,
                                                const bf16* __restrict__ W,
                                                const float* __restrict__ bias,
                                                float* __restrict__ pre)
{
    __shared__ __align__(16) bf16 lb[2][128 * LBPAD];
    int bid = blockIdx.x;
    int gt = bid / 6, mt = bid % 6;
    int t = gt % T_;
    int tid = threadIdx.x;
    int wave = tid >> 5, lane = tid & 31;
    int hi = (lane >> 4) & 1;                      // upper half-wave flag
    int m0 = mt * 128 + wave * 16;
    const bf16* arow = A + ((size_t)t * ROWS + m0 + (lane & 15)) * KTOT;
    const bf16* wsl = W + (size_t)gt * H_ * KTOT;

    // cooperative B staging: 2 threads per n-row, 32 bytes each
    int ln = tid >> 1, lhalf = tid & 1;
    const bf16* wrow = wsl + (size_t)ln * KTOT + lhalf * 16;
    int lds_off = ln * LBPAD + lhalf * 16;

    v8f acc[8] = {};

    // stage step 0 into lb[0]
    stage16(&lb[0][0] + lds_off, wrow);
    stage16(&lb[0][0] + lds_off + 8, wrow + 8);
    stage_commit();

    int cur = 0;
    for (int s = 0; s < 28; s++) {
        if (s < 27) {
            bf16* ld2 = &lb[cur ^ 1][0] + lds_off;
            const bf16* wsrc = wrow + (s + 1) * 32;
            stage16(ld2, wsrc);
            stage16(ld2 + 8, wsrc + 8);
        }
        // A fragment: lanes<16 take K {ks..ks+7, ks+16..ks+23}; lanes>=16 shifted by 8
        int kk = s * 32 + hi * 8;
        union { uint4 u[2]; v16bf v; } af;
        af.u[0] = *(const uint4*)(arow + kk);
        af.u[1] = *(const uint4*)(arow + kk + 16);
        const bf16* lbc = &lb[cur][0];
#pragma unroll
        for (int j = 0; j < 8; j++) {
            const bf16* bp = lbc + (j * 16 + (lane & 15)) * LBPAD + hi * 8;
            union { uint4 u[2]; v16bf v; } bfrag;
            bfrag.u[0] = *(const uint4*)(bp);
            bfrag.u[1] = *(const uint4*)(bp + 16);
            acc[j] = __builtin_amdgcn_wmma_f32_16x16x32_bf16(
                false, af.v, false, bfrag.v, (short)0, acc[j], false, false);
        }
        if (s < 27) {
            stage_commit();
            cur ^= 1;
        }
    }
    // epilogue: bias add + store f32
    const float* bv = bias + (size_t)gt * H_;
    float* pb = pre + (size_t)gt * ROWS * H_;
#pragma unroll
    for (int j = 0; j < 8; j++) {
        int n = j * 16 + (lane & 15);
        float bn = bv[n];
#pragma unroll
        for (int r = 0; r < 8; r++) {
            int row = m0 + r + hi * 8;
            pb[(size_t)row * H_ + n] = acc[j][r] + bn;
        }
    }
}

// ---------------- kernel 5: fused gate activations + cell update ----------------
__global__ __launch_bounds__(256) void fuse_cell(const float* __restrict__ pre,
                                                 const float* __restrict__ c_h,
                                                 const float* __restrict__ c_gt,
                                                 const float* __restrict__ c_gs,
                                                 float* __restrict__ o_h,
                                                 float* __restrict__ o_c,
                                                 bf16* __restrict__ hnbf)
{
    size_t i = (size_t)blockIdx.x * 256 + threadIdx.x;     // over B*T*NB*H
    if (i >= (size_t)B_ * T_ * NB * H_) return;
    int hh = (int)(i & (H_ - 1));
    int nb = (int)((i >> 7) % NB);
    int t  = (int)((i / ((size_t)H_ * NB)) % T_);
    int b  = (int)(i / ((size_t)H_ * NB * T_));
    int row = b * NB + nb;
    size_t poff = ((size_t)t * ROWS + row) * H_ + hh;      // inside one gate slab
    const size_t gstride = (size_t)T_ * ROWS * H_;
    float pg[9];
#pragma unroll
    for (int g = 0; g < 9; g++) pg[g] = pre[(size_t)g * gstride + poff];
    float iv  = sigf(pg[0]);
    float lt  = sigf(pg[1]);
    float ft  = sigf(pg[2]);
    float rt  = sigf(pg[3]);
    float sg  = sigf(pg[4]);
    float gtg = sigf(pg[5]);
    float gsg = sigf(pg[6]);
    float ov  = sigf(pg[7]);
    float cand = tanhf(pg[8]);

    size_t cbase = ((size_t)(b * T_ + t) * NB + nb) * H_ + hh;
    float cc  = c_h[cbase];
    float ctb = (t > 0)      ? c_h[cbase - (size_t)NB * H_] : 0.f;
    float cta = (t < T_ - 1) ? c_h[cbase + (size_t)NB * H_] : 0.f;
    float cs  = ((nb > 0)      ? c_h[cbase - H_] : 0.f)
              + ((nb < NB - 1) ? c_h[cbase + H_] : 0.f);
    float cg_t = c_gt[(size_t)row * H_ + hh];
    float cg_s = c_gs[((size_t)b * T_ + t) * H_ + hh];

    float cn = lt * ctb + ft * cc + rt * cta + sg * cs
             + gtg * cg_t + gsg * cg_s + iv * cand;
    float hn = ov * tanhf(cn);
    o_h[cbase] = hn;
    o_c[cbase] = cn;
    hnbf[poff] = (bf16)hn;       // [T][768][128] bf16 copy for the recurrence
}

// ---------------- kernel 6: mean over bones ----------------
__global__ __launch_bounds__(256) void mean_bones(const float* __restrict__ hn,
                                                  float* __restrict__ hm)
{
    int i = blockIdx.x * 256 + threadIdx.x;     // over B*T*H
    if (i >= B_ * T_ * H_) return;
    int hh = i & (H_ - 1);
    int t = (i >> 7) % T_;
    int b = i / (H_ * T_);
    float s = 0.f;
#pragma unroll 4
    for (int nb = 0; nb < NB; nb++)
        s += hn[((size_t)(b * T_ + t) * NB + nb) * H_ + hh];
    hm[i] = s * (1.f / NB);
}

// ---------------- kernel 7: g_s global spatial update (small, VALU) ----------------
__global__ __launch_bounds__(256) void gs_update(const float* __restrict__ hm,
                                                 const float* __restrict__ gs,
                                                 const float* __restrict__ cgs,
                                                 const float* __restrict__ Wgs,
                                                 const float* __restrict__ Zgs,
                                                 const float* __restrict__ bgs,
                                                 float* __restrict__ o_gs,
                                                 float* __restrict__ o_cgs)
{
    int i = blockIdx.x * 256 + threadIdx.x;     // over B*T*H
    if (i >= B_ * T_ * H_) return;
    int ho = i & (H_ - 1);
    int t = (i >> 7) % T_;
    int b = i / (H_ * T_);
    const float* hv = hm + ((size_t)b * T_ + t) * H_;
    const float* gv = gs + ((size_t)b * T_ + t) * H_;
    float pk[3];
#pragma unroll
    for (int k = 0; k < 3; k++) {
        const float* W = Wgs + ((size_t)(k * T_ + t) * H_) * H_ + ho;
        const float* Z = Zgs + ((size_t)(k * T_ + t) * H_) * H_ + ho;
        float s = bgs[(size_t)(k * T_ + t) * H_ + ho];
        for (int h2 = 0; h2 < H_; h2++)
            s += hv[h2] * W[(size_t)h2 * H_] + gv[h2] * Z[(size_t)h2 * H_];
        pk[k] = s;
    }
    float f  = sigf(pk[0]);
    float gc = tanhf(pk[1]);
    float og = sigf(pk[2]);
    float cold = cgs[i];
    float cn = f * cold + (1.f - f) * gc;
    o_cgs[i] = cn;
    o_gs[i]  = og * tanhf(cn);
}

// ---------------- kernel 8: one step of the g_t recurrence (WMMA, K=256 = [h_t | g]) ----------------
__global__ __launch_bounds__(256) void gt_step(const bf16* __restrict__ hnbf,
                                               const bf16* __restrict__ WgtT,
                                               const float* __restrict__ bgt,
                                               bf16* __restrict__ gbf,
                                               float* __restrict__ cbuf,
                                               float* __restrict__ o_g,
                                               float* __restrict__ o_c,
                                               int t, int is_last)
{
    __shared__ __align__(16) bf16 lb[2][128 * LBPAD];
    int tid = threadIdx.x;
    int wave = tid >> 5, lane = tid & 31;
    int hi = (lane >> 4) & 1;
    int m0 = blockIdx.x * 128 + wave * 16;
    int arow_r = m0 + (lane & 15);
    const bf16* hrow = hnbf + ((size_t)t * ROWS + arow_r) * H_;
    const bf16* grow = gbf + (size_t)arow_r * H_;
    int ln = tid >> 1, lhalf = tid & 1;
    int lds_off = ln * LBPAD + lhalf * 16;

    v8f acc[3][8] = {};

    for (int k3 = 0; k3 < 3; k3++) {
        const bf16* wsl = WgtT + (size_t)(k3 * T_ + t) * H_ * 256;
        const bf16* wrow = wsl + (size_t)ln * 256 + lhalf * 16;
        __syncthreads();   // previous gate's LDS reads done before overwrite
        stage16(&lb[0][0] + lds_off, wrow);
        stage16(&lb[0][0] + lds_off + 8, wrow + 8);
        stage_commit();
        int cur = 0;
        for (int s = 0; s < 8; s++) {
            if (s < 7) {
                bf16* ld2 = &lb[cur ^ 1][0] + lds_off;
                const bf16* wsrc = wrow + (s + 1) * 32;
                stage16(ld2, wsrc);
                stage16(ld2 + 8, wsrc + 8);
            }
            int kk = (s & 3) * 32 + hi * 8;
            const bf16* asrc = (s < 4) ? hrow : grow;
            union { uint4 u[2]; v16bf v; } af;
            af.u[0] = *(const uint4*)(asrc + kk);
            af.u[1] = *(const uint4*)(asrc + kk + 16);
            const bf16* lbc = &lb[cur][0];
#pragma unroll
            for (int j = 0; j < 8; j++) {
                const bf16* bp = lbc + (j * 16 + (lane & 15)) * LBPAD + hi * 8;
                union { uint4 u[2]; v16bf v; } bfrag;
                bfrag.u[0] = *(const uint4*)(bp);
                bfrag.u[1] = *(const uint4*)(bp + 16);
                acc[k3][j] = __builtin_amdgcn_wmma_f32_16x16x32_bf16(
                    false, af.v, false, bfrag.v, (short)0, acc[k3][j], false, false);
            }
            if (s < 7) {
                stage_commit();
                cur ^= 1;
            }
        }
    }
    // epilogue: fused LSTM update (rows are wave-private -> no races)
    const float* b0 = bgt + (size_t)(0 * T_ + t) * H_;
    const float* b1 = bgt + (size_t)(1 * T_ + t) * H_;
    const float* b2 = bgt + (size_t)(2 * T_ + t) * H_;
#pragma unroll
    for (int j = 0; j < 8; j++) {
        int n = j * 16 + (lane & 15);
        float bb0 = b0[n], bb1 = b1[n], bb2 = b2[n];
#pragma unroll
        for (int r = 0; r < 8; r++) {
            int row = m0 + r + hi * 8;
            size_t off = (size_t)row * H_ + n;
            float f  = sigf(acc[0][j][r] + bb0);
            float gc = tanhf(acc[1][j][r] + bb1);
            float og = sigf(acc[2][j][r] + bb2);
            float c = cbuf[off];
            float cn = f * c + (1.f - f) * gc;
            float gn = og * tanhf(cn);
            cbuf[off] = cn;
            gbf[off] = (bf16)gn;
            if (is_last) { o_g[off] = gn; o_c[off] = cn; }
        }
    }
}

// ---------------- launcher ----------------
extern "C" void kernel_launch(void* const* d_in, const int* in_sizes, int n_in,
                              void* d_out, int out_size, void* d_ws, size_t ws_size,
                              hipStream_t stream)
{
    (void)in_sizes; (void)n_in; (void)out_size; (void)ws_size;
    const float* h    = (const float*)d_in[0];
    const float* c_h  = (const float*)d_in[1];
    const float* p    = (const float*)d_in[2];
    const float* g_t  = (const float*)d_in[3];
    const float* c_gt = (const float*)d_in[4];
    const float* g_s  = (const float*)d_in[5];
    const float* c_gs = (const float*)d_in[6];
    const float* U    = (const float*)d_in[7];
    const float* Wt   = (const float*)d_in[8];
    const float* Ws   = (const float*)d_in[9];
    const float* Zt   = (const float*)d_in[10];
    const float* Zs   = (const float*)d_in[11];
    const float* bb   = (const float*)d_in[12];
    const float* Wgt  = (const float*)d_in[13];
    const float* Zgt  = (const float*)d_in[14];
    const float* bgt  = (const float*)d_in[15];
    const float* Wgs  = (const float*)d_in[16];
    const float* Zgs  = (const float*)d_in[17];
    const float* bgs  = (const float*)d_in[18];

    char* w = (char*)d_ws;
    bf16*  A    = (bf16*)w;  w += (size_t)T_ * ROWS * KTOT * 2;        // 67.4 MB
    bf16*  WpT  = (bf16*)w;  w += (size_t)GT_ * H_ * KTOT * 2;         // 101.2 MB
    float* pre  = (float*)w; w += (size_t)GT_ * ROWS * H_ * 4;         // 173.4 MB
    bf16*  hnbf = (bf16*)w;  w += (size_t)T_ * ROWS * H_ * 2;          // 9.6 MB
    bf16*  WgtT = (bf16*)w;  w += (size_t)3 * T_ * H_ * 256 * 2;       // 9.6 MB
    bf16*  gbf  = (bf16*)w;  w += (size_t)ROWS * H_ * 2;
    float* cbuf = (float*)w; w += (size_t)ROWS * H_ * 4;
    float* hm   = (float*)w; w += (size_t)B_ * T_ * H_ * 4;

    float* out   = (float*)d_out;
    float* o_h   = out;
    float* o_c   = o_h + (size_t)B_ * T_ * NB * H_;
    float* o_gt  = o_c + (size_t)B_ * T_ * NB * H_;
    float* o_cgt = o_gt + (size_t)ROWS * H_;
    float* o_gs  = o_cgt + (size_t)ROWS * H_;
    float* o_cgs = o_gs + (size_t)B_ * T_ * H_;

    dim3 tb(256);
    dim3 tt(32, 8);
    pack_A<<<(T_ * ROWS * H_ + 255) / 256, tb, 0, stream>>>(h, p, g_t, g_s, A);
    // weight packs: k-offsets 0 | 128..511 | 512 | 640 | 768 cover KTOT=896
    pack_wT<<<dim3(4, 4, GT_),  tt, 0, stream>>>(U,   WpT, 128, KTOT, 0);
    pack_wT<<<dim3(12, 4, GT_), tt, 0, stream>>>(Wt,  WpT, 384, KTOT, 128);
    pack_wT<<<dim3(4, 4, GT_),  tt, 0, stream>>>(Ws,  WpT, 128, KTOT, 512);
    pack_wT<<<dim3(4, 4, GT_),  tt, 0, stream>>>(Zt,  WpT, 128, KTOT, 640);
    pack_wT<<<dim3(4, 4, GT_),  tt, 0, stream>>>(Zs,  WpT, 128, KTOT, 768);
    pack_wT<<<dim3(4, 4, 3 * T_), tt, 0, stream>>>(Wgt, WgtT, 128, 256, 0);
    pack_wT<<<dim3(4, 4, 3 * T_), tt, 0, stream>>>(Zgt, WgtT, 128, 256, 128);
    init_gt<<<(ROWS * H_ + 255) / 256, tb, 0, stream>>>(g_t, c_gt, gbf, cbuf);

    gemm_pre<<<GT_ * 6, tb, 0, stream>>>(A, WpT, bb, pre);
    fuse_cell<<<(T_ * ROWS * H_ + 255) / 256, tb, 0, stream>>>(pre, c_h, c_gt, c_gs,
                                                               o_h, o_c, hnbf);
    mean_bones<<<(B_ * T_ * H_ + 255) / 256, tb, 0, stream>>>(o_h, hm);
    gs_update<<<(B_ * T_ * H_ + 255) / 256, tb, 0, stream>>>(hm, g_s, c_gs,
                                                             Wgs, Zgs, bgs, o_gs, o_cgs);
    for (int t = 0; t < T_; t++)
        gt_step<<<6, tb, 0, stream>>>(hnbf, WgtT, bgt, gbf, cbuf,
                                      o_gt, o_cgt, t, (t == T_ - 1) ? 1 : 0);
}
